// ST_transor_61735859913596
// MI455X (gfx1250) — matrix operand order repaired and checked by reading, not compiled, not run on hardware
//
#include <hip/hip_runtime.h>
#include <math.h>

// ---------------- problem constants ----------------
constexpr int kT  = 24;
constexpr int kN  = 10000;
constexpr int kE  = 320000;
constexpr int kP  = 20000;
constexpr int kD  = 64;
constexpr int kF  = 32;
constexpr int kXs = kF + 2;   // 34 floats per (t,n) row of x
constexpr int kCin = kD + kF; // 96

// ---------------- CDNA5 WMMA types ----------------
typedef __bf16 bf16_t;
typedef bf16_t v16bf __attribute__((ext_vector_type(16)));
typedef float  v8f   __attribute__((ext_vector_type(8)));

__device__ __forceinline__ unsigned short f2bf(float f) {
  unsigned u = __float_as_uint(f);
  u += 0x7FFFu + ((u >> 16) & 1u);      // round-to-nearest-even
  return (unsigned short)(u >> 16);
}

// ---------------- utility kernels ----------------
__global__ void k_zero_f(float* p, int n) {
  int i = blockIdx.x * blockDim.x + threadIdx.x;
  if (i < n) p[i] = 0.f;
}
__global__ void k_zero_i(int* p, int n) {
  int i = blockIdx.x * blockDim.x + threadIdx.x;
  if (i < n) p[i] = 0;
}

__global__ void k_count(const int* __restrict__ src, const int* __restrict__ dst,
                        int* __restrict__ cA, int* __restrict__ cB) {
  int e = blockIdx.x * blockDim.x + threadIdx.x;
  if (e < kE) {
    atomicAdd(&cA[src[e]], 1);
    atomicAdd(&cB[dst[e]], 1);
  }
}

__global__ void k_norm(const int* __restrict__ cA, const int* __restrict__ cB,
                       float* __restrict__ on, float* __restrict__ inn) {
  int n = blockIdx.x * blockDim.x + threadIdx.x;
  if (n < kN) {
    on[n]  = rsqrtf(fmaxf((float)cA[n], 1.f));
    inn[n] = rsqrtf(fmaxf((float)cB[n], 1.f));
  }
}

// ---- exclusive prefix scan of deg_in -> CSR offsets ----
__global__ void k_scan1(const int* __restrict__ cnt, int* __restrict__ offs,
                        int* __restrict__ bsum) {
  __shared__ int sm[256];
  int idx = blockIdx.x * 256 + threadIdx.x;
  int v = (idx < kN) ? cnt[idx] : 0;
  sm[threadIdx.x] = v;
  __syncthreads();
  for (int s = 1; s < 256; s <<= 1) {
    int t = (threadIdx.x >= s) ? sm[threadIdx.x - s] : 0;
    __syncthreads();
    sm[threadIdx.x] += t;
    __syncthreads();
  }
  if (idx < kN) offs[idx] = sm[threadIdx.x] - v;   // exclusive within block
  if (threadIdx.x == 255) bsum[blockIdx.x] = sm[255];
}
__global__ void k_scan2(int* bsum, int nb) {
  if (blockIdx.x == 0 && threadIdx.x == 0) {
    int run = 0;
    for (int i = 0; i < nb; ++i) { int t = bsum[i]; bsum[i] = run; run += t; }
  }
}
__global__ void k_scan3(int* __restrict__ offs, const int* __restrict__ bsum,
                        int* __restrict__ cursor) {
  int idx = blockIdx.x * 256 + threadIdx.x;
  if (idx < kN) {
    int v = offs[idx] + bsum[blockIdx.x];
    offs[idx] = v;
    cursor[idx] = v;
  }
  if (idx == 0) offs[kN] = kE;
}

__global__ void k_fill_csr(const int* __restrict__ src, const int* __restrict__ dst,
                           const float* __restrict__ on, const float* __restrict__ inn,
                           int* __restrict__ cursor, int* __restrict__ csrc,
                           float* __restrict__ ccoef) {
  int e = blockIdx.x * blockDim.x + threadIdx.x;
  if (e < kE) {
    int d = dst[e], s = src[e];
    int pos = atomicAdd(&cursor[d], 1);
    csrc[pos]  = s;
    ccoef[pos] = on[s] * inn[d];   // fold both norms into one edge coeff
  }
}

// ---- pack weight matrix (K x Mout, f32 row-major) into bf16 B-fragments ----
// frag id = (kc*ntn + nt)*32 + lane ; 16 contiguous bf16 per lane.
// B element i (0..15) of lane l: k = kc*32 + ((l&16)?16:0) + i ; col = nt*16 + (l&15)
__global__ void k_pack(const float* __restrict__ W, int K, int Mout,
                       unsigned short* __restrict__ o) {
  int ntn = Mout >> 4;
  int total = (K >> 5) * ntn * 32;
  int tid = blockIdx.x * blockDim.x + threadIdx.x;
  if (tid >= total) return;
  int lane = tid & 31;
  int t2 = tid >> 5;
  int nt = t2 % ntn;
  int kc = t2 / ntn;
  int col = nt * 16 + (lane & 15);
  int kb = (lane & 16) ? 16 : 0;
  unsigned short* d16 = o + (size_t)tid * 16;
  for (int i = 0; i < 16; ++i) {
    int k = kc * 32 + kb + i;
    d16[i] = f2bf(W[(size_t)k * Mout + col]);
  }
}

// ---- concat builders (f32 outputs; agg converts to bf16 downstream) ----
__global__ void k_concat0(const float* __restrict__ h, const float* __restrict__ xt,
                          float* __restrict__ C) {
  int idx = blockIdx.x * blockDim.x + threadIdx.x;
  if (idx >= kN * kCin) return;
  int n = idx / kCin, j = idx - n * kCin;
  C[idx] = (j < kD) ? h[(size_t)n * kD + j] : xt[(size_t)n * kXs + (j - kD)];
}
__global__ void k_concat_r(const float* __restrict__ h, const float* __restrict__ ru,
                           const float* __restrict__ xt, float* __restrict__ C) {
  int idx = blockIdx.x * blockDim.x + threadIdx.x;
  if (idx >= kN * kCin) return;
  int n = idx / kCin, j = idx - n * kCin;
  C[idx] = (j < kD) ? h[(size_t)n * kD + j] * ru[(size_t)n * (2 * kD) + j]
                    : xt[(size_t)n * kXs + (j - kD)];
}

// ---- CSR gather-aggregate: out[n] = sum_{e in->n} Hin[src(e)] * coeff(e) ----
// one wave per node; register accumulation; no atomics; L2-resident gathers.
// Output written as bf16 row-major: the GEMM consumes it as A fragments with
// zero conversion work.
template <int DIM>
__global__ void k_agg(const float* __restrict__ Hin, const int* __restrict__ csrc,
                      const float* __restrict__ ccoef, const int* __restrict__ offs,
                      unsigned short* __restrict__ out) {
  int node = blockIdx.x * (blockDim.x >> 5) + (threadIdx.x >> 5);
  if (node >= kN) return;                 // uniform per wave -> whole wave exits
  int lane = threadIdx.x & 31;
  constexpr int CPL = DIM / 32;
  float acc[CPL];
#pragma unroll
  for (int j = 0; j < CPL; ++j) acc[j] = 0.f;
  int beg = offs[node], end = offs[node + 1];
  for (int i = beg; i < end; ++i) {
    int s = csrc[i];
    float c = ccoef[i];
    const float* hr = Hin + (size_t)s * DIM;
    if (i + 1 < end)
      __builtin_prefetch(Hin + (size_t)csrc[i + 1] * DIM + lane, 0, 1);
#pragma unroll
    for (int j = 0; j < CPL; ++j) acc[j] += hr[lane + j * 32] * c;
  }
  unsigned short* op = out + (size_t)node * DIM;
#pragma unroll
  for (int j = 0; j < CPL; ++j) op[lane + j * 32] = f2bf(acc[j]);
}

// ---- bf16 WMMA GEMM: out = act(A(N x K, bf16) @ W(K x Mout) + b) ----
// One wave per 16-row stripe: loads each A fragment once per K-chunk and
// reuses it across NTN column tiles (independent WMMA accumulator chains).
// MODE: 0 relu, 1 sigmoid (-> ru buffer), 2 tanh + fused GRU update.
template <int MODE, int NTN, int K>
__global__ void k_gemm(const unsigned short* __restrict__ Abf,
                       const unsigned short* __restrict__ Wp,
                       const float* __restrict__ bias,
                       float* __restrict__ out,
                       const float* __restrict__ ru,
                       float* __restrict__ h,
                       float* __restrict__ hs_t) {
  constexpr int Mout = NTN * 16;
  constexpr int NKC = K >> 5;
  int tm = blockIdx.x * (blockDim.x >> 5) + (threadIdx.x >> 5);
  if (tm >= kN / 16) return;              // wave-uniform guard, EXEC stays all-1
  int lane = threadIdx.x & 31;
  int r16 = lane & 15;
  int kbA = (lane & 16) ? 16 : 0;         // contiguous k order used here: lanes>=16 hold K+16
  v8f acc[NTN] = {};
  const unsigned short* arow = Abf + (size_t)(tm * 16 + r16) * K + kbA;
#pragma unroll
  for (int kc = 0; kc < NKC; ++kc) {
    union Uf { uint4 u[2]; v16bf v; } ua;
    const uint4* ap = (const uint4*)(arow + kc * 32);
    ua.u[0] = ap[0];                      // k = kc*32 + kb   .. +7
    ua.u[1] = ap[1];                      // k = kc*32+8+kb   .. +7
#pragma unroll
    for (int t = 0; t < NTN; ++t) {
      union Uf ub;
      const uint4* wp4 =
          (const uint4*)(Wp + ((size_t)(kc * NTN + t) * 32 + lane) * 16);
      ub.u[0] = wp4[0];
      ub.u[1] = wp4[1];
      acc[t] = __builtin_amdgcn_wmma_f32_16x16x32_bf16(
          false, ua.v, false, ub.v, (short)0, acc[t], false, false);
    }
  }
  int rbase = tm * 16 + ((lane & 16) ? 8 : 0);   // C/D layout
#pragma unroll
  for (int t = 0; t < NTN; ++t) {
    int col = t * 16 + r16;
    float bc = bias[col];
#pragma unroll
    for (int v = 0; v < 8; ++v) {
      int row = rbase + v;
      float val = acc[t][v] + bc;
      if (MODE == 0) {
        out[(size_t)row * Mout + col] = fmaxf(val, 0.f);
      } else if (MODE == 1) {
        out[(size_t)row * Mout + col] = 1.f / (1.f + __expf(-val));
      } else {
        float hn = tanhf(val);
        float z  = ru[(size_t)row * (2 * kD) + kD + col];
        float ho = h[(size_t)row * kD + col];
        float nh = z * ho + (1.f - z) * hn;
        h[(size_t)row * kD + col]    = nh;
        hs_t[(size_t)row * kD + col] = nh;
      }
    }
  }
}

// ---- prediction head: out[p,t] = tanh([hs[t,o], hs[t,d], dis] . pW + pb) ----
__global__ void k_pred(const float* __restrict__ hs, const int* __restrict__ ti,
                       const float* __restrict__ dis, const float* __restrict__ pW,
                       const float* __restrict__ pb, float* __restrict__ out) {
  int p = blockIdx.x * blockDim.x + threadIdx.x;
  if (p >= kP) return;
  int o = ti[p], dd = ti[kP + p];
  float dval = dis[(size_t)o * kN + dd];
  float base = pb[0] + dval * pW[2 * kD];
  const float4* w4 = (const float4*)pW;
  for (int t = 0; t < kT; ++t) {
    const float4* ho = (const float4*)(hs + ((size_t)t * kN + o) * kD);
    const float4* hd = (const float4*)(hs + ((size_t)t * kN + dd) * kD);
    float s = base;
#pragma unroll
    for (int k = 0; k < 16; ++k) {
      float4 a = ho[k], wa = w4[k];
      float4 b = hd[k], wb = w4[16 + k];
      s += a.x * wa.x + a.y * wa.y + a.z * wa.z + a.w * wa.w;
      s += b.x * wb.x + b.y * wb.y + b.z * wb.z + b.w * wb.w;
    }
    out[(size_t)p * kT + t] = tanhf(s);
  }
}

// ---------------- host orchestration ----------------
extern "C" void kernel_launch(void* const* d_in, const int* in_sizes, int n_in,
                              void* d_out, int out_size, void* d_ws, size_t ws_size,
                              hipStream_t stream) {
  (void)in_sizes; (void)n_in; (void)out_size; (void)ws_size;

  const float* x   = (const float*)d_in[0];
  const float* dis = (const float*)d_in[1];
  const int*   src = (const int*)d_in[2];
  const int*   dst = (const int*)d_in[3];
  const int*   ti  = (const int*)d_in[4];
  const float* Wm[6]   = {(const float*)d_in[5],  (const float*)d_in[7],
                          (const float*)d_in[9],  (const float*)d_in[11],
                          (const float*)d_in[13], (const float*)d_in[15]};
  const float* Bias[6] = {(const float*)d_in[6],  (const float*)d_in[8],
                          (const float*)d_in[10], (const float*)d_in[12],
                          (const float*)d_in[14], (const float*)d_in[16]};
  const float* pW = (const float*)d_in[17];
  const float* pb = (const float*)d_in[18];
  float* out = (float*)d_out;

  const int Kdim[6] = {96, 64, 64, 96, 64, 64};
  const int Mdim[6] = {64, 64, 128, 64, 64, 64};

  // workspace carve-out (all offsets 256B aligned)
  char* ws = (char*)d_ws;
  size_t off = 0;
  auto take = [&](size_t bytes) -> char* {
    char* p = ws + off;
    off = (off + bytes + 255) & ~(size_t)255;
    return p;
  };
  float* outn   = (float*)take((size_t)kN * 4);
  float* innm   = (float*)take((size_t)kN * 4);
  int*   cntA   = (int*)take((size_t)kN * 4);
  int*   cntB   = (int*)take((size_t)kN * 4);
  int*   csroff = (int*)take((size_t)(kN + 1) * 4);
  int*   cursor = (int*)take((size_t)kN * 4);
  int*   bsum   = (int*)take(64 * 4);
  int*   csrc   = (int*)take((size_t)kE * 4);
  float* ccoef  = (float*)take((size_t)kE * 4);
  float* hbuf   = (float*)take((size_t)kN * kD * 4);
  float* Cbuf   = (float*)take((size_t)kN * kCin * 4);
  unsigned short* Abuf = (unsigned short*)take((size_t)kN * kCin * 2); // bf16 A
  float* Bbuf   = (float*)take((size_t)kN * kD * 4);
  float* rubuf  = (float*)take((size_t)kN * 2 * kD * 4);
  float* hs     = (float*)take((size_t)kT * kN * kD * 4);
  unsigned short* wp[6];
  for (int i = 0; i < 6; ++i)
    wp[i] = (unsigned short*)take((size_t)Kdim[i] * Mdim[i] * 2);

  const int TB = 256;
  auto nblk = [](int n, int tb) { return (n + tb - 1) / tb; };

  // ---- graph preprocessing ----
  k_zero_i<<<nblk(kN, TB), TB, 0, stream>>>(cntA, kN);
  k_zero_i<<<nblk(kN, TB), TB, 0, stream>>>(cntB, kN);
  k_zero_f<<<nblk(kN * kD, TB), TB, 0, stream>>>(hbuf, kN * kD);
  k_count<<<nblk(kE, TB), TB, 0, stream>>>(src, dst, cntA, cntB);
  k_norm<<<nblk(kN, TB), TB, 0, stream>>>(cntA, cntB, outn, innm);
  int nb = (kN + 255) / 256;
  k_scan1<<<nb, 256, 0, stream>>>(cntB, csroff, bsum);
  k_scan2<<<1, 32, 0, stream>>>(bsum, nb);
  k_scan3<<<nb, 256, 0, stream>>>(csroff, bsum, cursor);
  k_fill_csr<<<nblk(kE, TB), TB, 0, stream>>>(src, dst, outn, innm, cursor, csrc, ccoef);
  for (int i = 0; i < 6; ++i) {
    int tot = (Kdim[i] >> 5) * (Mdim[i] >> 4) * 32;
    k_pack<<<nblk(tot, TB), TB, 0, stream>>>(Wm[i], Kdim[i], Mdim[i], wp[i]);
  }

  const int aggBlocks  = (kN + 7) / 8;           // 8 waves (nodes) per block
  const int gemmBlocks = (kN / 16 + 7) / 8;      // 8 row-stripes per block

  // ---- temporal GRU scan ----
  for (int t = 0; t < kT; ++t) {
    const float* xt = x + (size_t)t * kN * kXs;
    float* hst = hs + (size_t)t * kN * kD;

    // ru = sigmoid(GNN([h, xt]))
    k_concat0<<<nblk(kN * kCin, TB), TB, 0, stream>>>(hbuf, xt, Cbuf);
    k_agg<96><<<aggBlocks, TB, 0, stream>>>(Cbuf, csrc, ccoef, csroff, Abuf);
    k_gemm<0, 4, 96><<<gemmBlocks, TB, 0, stream>>>(Abuf, wp[0], Bias[0],
                                                    Bbuf, rubuf, hbuf, hst);
    k_agg<64><<<aggBlocks, TB, 0, stream>>>(Bbuf, csrc, ccoef, csroff, Abuf);
    k_gemm<0, 4, 64><<<gemmBlocks, TB, 0, stream>>>(Abuf, wp[1], Bias[1],
                                                    Bbuf, rubuf, hbuf, hst);
    k_agg<64><<<aggBlocks, TB, 0, stream>>>(Bbuf, csrc, ccoef, csroff, Abuf);
    k_gemm<1, 8, 64><<<gemmBlocks, TB, 0, stream>>>(Abuf, wp[2], Bias[2],
                                                    rubuf, rubuf, hbuf, hst);

    // h_new = tanh(GNN([h*r, xt])) ; h = z*h + (1-z)*h_new (fused epilogue)
    k_concat_r<<<nblk(kN * kCin, TB), TB, 0, stream>>>(hbuf, rubuf, xt, Cbuf);
    k_agg<96><<<aggBlocks, TB, 0, stream>>>(Cbuf, csrc, ccoef, csroff, Abuf);
    k_gemm<0, 4, 96><<<gemmBlocks, TB, 0, stream>>>(Abuf, wp[3], Bias[3],
                                                    Bbuf, rubuf, hbuf, hst);
    k_agg<64><<<aggBlocks, TB, 0, stream>>>(Bbuf, csrc, ccoef, csroff, Abuf);
    k_gemm<0, 4, 64><<<gemmBlocks, TB, 0, stream>>>(Abuf, wp[4], Bias[4],
                                                    Bbuf, rubuf, hbuf, hst);
    k_agg<64><<<aggBlocks, TB, 0, stream>>>(Bbuf, csrc, ccoef, csroff, Abuf);
    k_gemm<2, 4, 64><<<gemmBlocks, TB, 0, stream>>>(Abuf, wp[5], Bias[5],
                                                    Bbuf, rubuf, hbuf, hst);
  }

  // ---- prediction head ----
  k_pred<<<nblk(kP, TB), TB, 0, stream>>>(hs, ti, dis, pW, pb, out);
}
